// PYG_GAT_1752346657316
// MI455X (gfx1250) — matrix operand order repaired and checked by reading, not compiled, not run on hardware
//
#include <hip/hip_runtime.h>
#include <math.h>

// ---------------- problem constants (from reference) ----------------
#define NN      50000
#define IN_F    128
#define HID     64
#define H1      4
#define OUT_F   32
#define H2      8
#define EE      800000
#define ET      (EE + NN)          // edges + self loops = 850000
#define F1      (H1 * HID)         // 256
#define F2      (H2 * OUT_F)       // 256
#define NEG_SLOPE 0.2f

typedef __attribute__((ext_vector_type(16))) _Float16 v16h;
typedef __attribute__((ext_vector_type(8)))  _Float16 v8h;
typedef __attribute__((ext_vector_type(8)))  float    v8f;

// ---------------- small helpers ----------------
__global__ void fill_f32(float* p, float v, long long n) {
    long long i = blockIdx.x * (long long)blockDim.x + threadIdx.x;
    if (i < n) p[i] = v;
}

__global__ void cvt_f32_f16(const float* __restrict__ src, _Float16* __restrict__ dst, long long n) {
    long long i = blockIdx.x * (long long)blockDim.x + threadIdx.x;
    if (i < n) dst[i] = (_Float16)src[i];
}

// W: [K, Nc] row-major f32  ->  Wt: [Nc, K] row-major f16 (i.e. column-major W)
__global__ void transpose_cvt(const float* __restrict__ W, _Float16* __restrict__ Wt, int K, int Nc) {
    long long i = blockIdx.x * (long long)blockDim.x + threadIdx.x;
    if (i < (long long)K * Nc) {
        int k = (int)(i / Nc), n = (int)(i % Nc);
        Wt[(long long)n * K + k] = (_Float16)W[i];
    }
}

__device__ inline void atomicMaxF(float* addr, float val) {
    if (val >= 0.0f) atomicMax((int*)addr, __float_as_int(val));
    else             atomicMin((unsigned int*)addr, __float_as_uint(val));
}

// ---------------- WMMA GEMM: C[M,Nc] = A[M,K] * B[K,Nc] ----------------
// A row-major f16, B given as Bt (col-major, [Nc][K]) f16, C row-major f32.
// One wave computes a 16x64 strip. M%16==0, K%32==0, Nc%64==0 (exact grids,
// so EXEC is all-ones for every WMMA as the ISA requires).
__device__ inline v16h load_a_frag(const _Float16* __restrict__ A, int K, int row, int kk, int lane) {
    int m = lane & 15, half = lane >> 4;
    const _Float16* p = A + (long long)(row + m) * K + kk + 8 * half;
    v8h lo = *(const v8h*)p;
    v8h hi = *(const v8h*)(p + 16);
    return __builtin_shufflevector(lo, hi, 0,1,2,3,4,5,6,7,8,9,10,11,12,13,14,15);
}

__device__ inline v16h load_b_frag(const _Float16* __restrict__ Bt, int K, int col0, int kk, int lane) {
    int n = lane & 15, half = lane >> 4;
    const _Float16* p = Bt + (long long)(col0 + n) * K + kk + 8 * half;
    v8h lo = *(const v8h*)p;
    v8h hi = *(const v8h*)(p + 16);
    return __builtin_shufflevector(lo, hi, 0,1,2,3,4,5,6,7,8,9,10,11,12,13,14,15);
}

__global__ void gemm_wmma_f16(const _Float16* __restrict__ A, const _Float16* __restrict__ Bt,
                              float* __restrict__ C, int M, int K, int Nc) {
    int lane = threadIdx.x & 31;
    int wave = (int)((blockIdx.x * (long long)blockDim.x + threadIdx.x) >> 5);
    int ngrp = Nc >> 6;                       // number of 64-wide column groups
    int mt = wave / ngrp, ng = wave % ngrp;
    if (mt * 16 >= M) return;                 // never taken with exact grid
    int row = mt * 16, col = ng * 64;

    v8f acc0 = {}, acc1 = {}, acc2 = {}, acc3 = {};
    for (int kk = 0; kk < K; kk += 32) {
        v16h a   = load_a_frag(A, K, row, kk, lane);
        v16h b0  = load_b_frag(Bt, K, col +  0, kk, lane);
        v16h b1v = load_b_frag(Bt, K, col + 16, kk, lane);
        v16h b2v = load_b_frag(Bt, K, col + 32, kk, lane);
        v16h b3v = load_b_frag(Bt, K, col + 48, kk, lane);
        acc0 = __builtin_amdgcn_wmma_f32_16x16x32_f16(false, a, false, b0,  (short)0, acc0, false, false);
        acc1 = __builtin_amdgcn_wmma_f32_16x16x32_f16(false, a, false, b1v, (short)0, acc1, false, false);
        acc2 = __builtin_amdgcn_wmma_f32_16x16x32_f16(false, a, false, b2v, (short)0, acc2, false, false);
        acc3 = __builtin_amdgcn_wmma_f32_16x16x32_f16(false, a, false, b3v, (short)0, acc3, false, false);
    }
    int n = lane & 15, half = lane >> 4;
    float* cbase = C + (long long)(row + 8 * half) * Nc + col + n;
#pragma unroll
    for (int r = 0; r < 8; r++) {
        cbase[(long long)r * Nc +  0] = acc0[r];
        cbase[(long long)r * Nc + 16] = acc1[r];
        cbase[(long long)r * Nc + 32] = acc2[r];
        cbase[(long long)r * Nc + 48] = acc3[r];
    }
}

// ---------------- attention scores: a_s[n,h] = sum_c h[n,h,c]*att_s[h,c] ----------------
template <int H, int C>
__global__ void scores_kernel(const float* __restrict__ hfeat, const float* __restrict__ att_s,
                              const float* __restrict__ att_d, float* __restrict__ as_,
                              float* __restrict__ ad_) {
    int lane = threadIdx.x & 31;
    long long wave = (blockIdx.x * (long long)blockDim.x + threadIdx.x) >> 5;
    if (wave >= (long long)NN * H) return;
    int n = (int)(wave / H), h = (int)(wave % H);
    const float* hp  = hfeat + (long long)n * H * C + (long long)h * C;
    const float* asp = att_s + (long long)h * C;
    const float* adp = att_d + (long long)h * C;
    float ss = 0.f, sd = 0.f;
#pragma unroll
    for (int c0 = 0; c0 < C; c0 += 32) {
        int c = c0 + lane;
        float v = hp[c];
        ss += v * asp[c];
        sd += v * adp[c];
    }
#pragma unroll
    for (int off = 16; off; off >>= 1) { ss += __shfl_xor(ss, off, 32); sd += __shfl_xor(sd, off, 32); }
    if (lane == 0) { as_[wave] = ss; ad_[wave] = sd; }
}

// ---------------- edge pass A: logits + segment max ----------------
template <int H>
__global__ void edge_logits(const int* __restrict__ ei, const float* __restrict__ as_,
                            const float* __restrict__ ad_, float* __restrict__ ebuf,
                            float* __restrict__ emax) {
    long long tid = blockIdx.x * (long long)blockDim.x + threadIdx.x;
    if (tid >= (long long)ET * H) return;
    int e = (int)(tid / H), h = (int)(tid % H);
    int s, d;
    if (e < EE) { s = ei[e]; d = ei[EE + e]; } else { s = d = e - EE; }
    float v = as_[(long long)s * H + h] + ad_[(long long)d * H + h];
    v = v > 0.f ? v : NEG_SLOPE * v;
    ebuf[tid] = v;
    atomicMaxF(&emax[(long long)d * H + h], v);
}

// ---------------- edge pass B: exp + segment sum ----------------
template <int H>
__global__ void edge_exp(const int* __restrict__ ei, float* __restrict__ ebuf,
                         const float* __restrict__ emax, float* __restrict__ denom) {
    long long tid = blockIdx.x * (long long)blockDim.x + threadIdx.x;
    if (tid >= (long long)ET * H) return;
    int e = (int)(tid / H), h = (int)(tid % H);
    int d = (e < EE) ? ei[EE + e] : (e - EE);
    float ex = expf(ebuf[tid] - emax[(long long)d * H + h]);
    ebuf[tid] = ex;
    atomicAdd(&denom[(long long)d * H + h], ex);
}

// ---------------- edge pass C: aggregate h[src]*alpha into dst (wave per edge) ----------------
// alpha[h] computed once per wave (lanes 0..H-1), broadcast via shuffle; the
// feature loop is fully unrolled with stride-32 accesses (one coalesced 128B
// request per iteration) and one global f32 atomicAdd per element.
template <int H, int C>
__global__ void edge_aggregate(const int* __restrict__ ei, const float* __restrict__ hfeat,
                               const float* __restrict__ ebuf, const float* __restrict__ denom,
                               float* __restrict__ agg) {
    constexpr int F = H * C;
    int lane = threadIdx.x & 31;
    long long wave = (blockIdx.x * (long long)blockDim.x + threadIdx.x) >> 5;
    if (wave >= (long long)ET) return;
    int e = (int)wave;
    int s, d;
    if (e < EE) { s = ei[e]; d = ei[EE + e]; } else { s = d = e - EE; }

    float alpha_l = 0.f;
    if (lane < H)
        alpha_l = ebuf[(long long)e * H + lane] / (denom[(long long)d * H + lane] + 1e-16f);

    const float* hp = hfeat + (long long)s * F;
    float* ap = agg + (long long)d * F;
#pragma unroll
    for (int f0 = 0; f0 < F; f0 += 32) {
        int f = f0 + lane;
        float alpha = __shfl(alpha_l, f / C, 32);
        atomicAdd(&ap[f], hp[f] * alpha);
    }
}

// ---------------- layer-1 epilogue: +bias, ELU, convert to f16 for GEMM2 ----------------
__global__ void bias_elu_cvt(const float* __restrict__ agg, const float* __restrict__ bias,
                             _Float16* __restrict__ outh, long long n, int F) {
    long long i = blockIdx.x * (long long)blockDim.x + threadIdx.x;
    if (i >= n) return;
    float v = agg[i] + bias[i % F];
    v = v > 0.f ? v : (expf(v) - 1.f);
    outh[i] = (_Float16)v;
}

// ---------------- layer-2 epilogue: mean over heads + bias ----------------
__global__ void head_mean_bias(const float* __restrict__ agg, const float* __restrict__ bias,
                               float* __restrict__ out) {
    long long i = blockIdx.x * (long long)blockDim.x + threadIdx.x;
    if (i >= (long long)NN * OUT_F) return;
    int n = (int)(i / OUT_F), c = (int)(i % OUT_F);
    const float* ap = agg + (long long)n * F2 + c;
    float s = 0.f;
#pragma unroll
    for (int h = 0; h < H2; h++) s += ap[h * OUT_F];
    out[i] = s * (1.0f / H2) + bias[c];
}

// ---------------- host orchestration ----------------
static inline unsigned nblk(long long n, int b) { return (unsigned)((n + b - 1) / b); }

extern "C" void kernel_launch(void* const* d_in, const int* in_sizes, int n_in,
                              void* d_out, int out_size, void* d_ws, size_t ws_size,
                              hipStream_t stream) {
    const float* x       = (const float*)d_in[0];
    const int*   ei      = (const int*)  d_in[1];
    const float* W1      = (const float*)d_in[2];
    const float* att_s1  = (const float*)d_in[3];
    const float* att_d1  = (const float*)d_in[4];
    const float* b1      = (const float*)d_in[5];
    const float* W2      = (const float*)d_in[6];
    const float* att_s2  = (const float*)d_in[7];
    const float* att_d2  = (const float*)d_in[8];
    const float* b2      = (const float*)d_in[9];
    float* out = (float*)d_out;

    // workspace layout (256B aligned chunks)
    char* ws = (char*)d_ws;
    size_t off = 0;
    auto take = [&](size_t bytes) { char* p = ws + off; off = (off + bytes + 255) & ~(size_t)255; return p; };
    _Float16* xh   = (_Float16*)take((size_t)NN * IN_F * 2);     // x as f16
    _Float16* w1t  = (_Float16*)take((size_t)F1 * IN_F * 2);     // W1 col-major f16
    _Float16* w2t  = (_Float16*)take((size_t)F2 * F1 * 2);       // W2 col-major f16
    float*    h1   = (float*)   take((size_t)NN * F1 * 4);       // GEMM1 out
    float*    h2   = (float*)   take((size_t)NN * F2 * 4);       // GEMM2 out
    float*    agg  = (float*)   take((size_t)NN * F1 * 4);       // aggregation (reused L1/L2)
    _Float16* h2h  = (_Float16*)take((size_t)NN * F1 * 2);       // elu(layer1) as f16
    float*    as1  = (float*)   take((size_t)NN * H1 * 4);
    float*    ad1  = (float*)   take((size_t)NN * H1 * 4);
    float*    as2  = (float*)   take((size_t)NN * H2 * 4);
    float*    ad2  = (float*)   take((size_t)NN * H2 * 4);
    float*    ebuf = (float*)   take((size_t)ET * H2 * 4);       // edge logits (max H)
    float*    emax = (float*)   take((size_t)NN * H2 * 4);
    float*    den  = (float*)   take((size_t)NN * H2 * 4);
    (void)ws_size; (void)n_in; (void)in_sizes; (void)out_size;

    const int B = 256;

    // ---- precision conversions ----
    cvt_f32_f16<<<nblk((long long)NN * IN_F, B), B, 0, stream>>>(x, xh, (long long)NN * IN_F);
    transpose_cvt<<<nblk((long long)IN_F * F1, B), B, 0, stream>>>(W1, w1t, IN_F, F1);
    transpose_cvt<<<nblk((long long)F1 * F2, B), B, 0, stream>>>(W2, w2t, F1, F2);

    // ================= layer 1 =================
    // GEMM1: h1 = x @ W1  (waves = (N/16)*(F1/64) = 12500; 4 waves/block)
    gemm_wmma_f16<<<(NN / 16) * (F1 / 64) / 4, 128, 0, stream>>>(xh, w1t, h1, NN, IN_F, F1);

    scores_kernel<H1, HID><<<nblk((long long)NN * H1 * 32, B), B, 0, stream>>>(h1, att_s1, att_d1, as1, ad1);

    fill_f32<<<nblk((long long)NN * H1, B), B, 0, stream>>>(emax, -INFINITY, (long long)NN * H1);
    fill_f32<<<nblk((long long)NN * H1, B), B, 0, stream>>>(den, 0.f, (long long)NN * H1);
    fill_f32<<<nblk((long long)NN * F1, B), B, 0, stream>>>(agg, 0.f, (long long)NN * F1);

    edge_logits<H1>        <<<nblk((long long)ET * H1, B), B, 0, stream>>>(ei, as1, ad1, ebuf, emax);
    edge_exp<H1>           <<<nblk((long long)ET * H1, B), B, 0, stream>>>(ei, ebuf, emax, den);
    edge_aggregate<H1, HID><<<nblk((long long)ET * 32, B), B, 0, stream>>>(ei, h1, ebuf, den, agg);

    bias_elu_cvt<<<nblk((long long)NN * F1, B), B, 0, stream>>>(agg, b1, h2h, (long long)NN * F1, F1);

    // ================= layer 2 =================
    gemm_wmma_f16<<<(NN / 16) * (F2 / 64) / 4, 128, 0, stream>>>(h2h, w2t, h2, NN, F1, F2);

    scores_kernel<H2, OUT_F><<<nblk((long long)NN * H2 * 32, B), B, 0, stream>>>(h2, att_s2, att_d2, as2, ad2);

    fill_f32<<<nblk((long long)NN * H2, B), B, 0, stream>>>(emax, -INFINITY, (long long)NN * H2);
    fill_f32<<<nblk((long long)NN * H2, B), B, 0, stream>>>(den, 0.f, (long long)NN * H2);
    fill_f32<<<nblk((long long)NN * F2, B), B, 0, stream>>>(agg, 0.f, (long long)NN * F2);

    edge_logits<H2>          <<<nblk((long long)ET * H2, B), B, 0, stream>>>(ei, as2, ad2, ebuf, emax);
    edge_exp<H2>             <<<nblk((long long)ET * H2, B), B, 0, stream>>>(ei, ebuf, emax, den);
    edge_aggregate<H2, OUT_F><<<nblk((long long)ET * 32, B), B, 0, stream>>>(ei, h2, ebuf, den, agg);

    head_mean_bias<<<nblk((long long)NN * OUT_F, B), B, 0, stream>>>(agg, b2, out);
}
